// UnifiedEmitterPredictor_2508260901677
// MI455X (gfx1250) — matrix operand order repaired and checked by reading, not compiled, not run on hardware
//
#include <hip/hip_runtime.h>
#include <hip/hip_bf16.h>

// VQ-VAE forward for MI455X (gfx1250, wave32, WMMA + TDM).
// - All GEMM-like work (convs, conv-transposes, VQ distances) on
//   v_wmma_f32_16x16x32_f16 (f16 in, f32 accumulate).
// - Implicit-GEMM conv: K loop reordered (tap outer, Cin inner) -> inner loop
//   is add-only addressing (no div/mod). 4x M-blocking: one B (im2col)
//   fragment feeds 4 WMMAs.
// - VQ stages codebook tiles into LDS with tensor_load_to_lds (TDM), using the
//   descriptor pad feature (row stride 260 dwords) to kill LDS bank conflicts.

typedef __attribute__((ext_vector_type(16))) _Float16 v16h;
typedef __attribute__((ext_vector_type(8)))  float    v8f;
typedef __attribute__((ext_vector_type(4)))  unsigned int v4u;
typedef __attribute__((ext_vector_type(8)))  int      v8i;
typedef __attribute__((ext_vector_type(4)))  int      v4i;

#define BNF 0.99999500003749972f

#if defined(__has_builtin)
#  if __has_builtin(__builtin_amdgcn_tensor_load_to_lds)
#    define HAS_TDM 1
#  endif
#endif

union AFrag { v16h v; _Float16 h[16]; };
union CFrag { v8f  v; float     f[8];  };

// ---------------------------------------------------------------------------
// Implicit-GEMM conv / conv-transpose, WMMA, 64(Cout) x 16(pixels) per wave.
// Fragment layouts used here (wave32; C as in CDNA5 ISA 7.12.2):
//   A 16x32 f16 : lane row = lane&15 ; half i: K = i + 16*(lane>=16) (contiguous)
//   B 32x16 f16 : lane col = lane&15 ; half i: K = i + 16*(lane>=16) (same order as A; not ISA table order)
//   C 16x16 f32 : lane col = lane&15 ; vgpr j: row = j + 8*(lane>=16)
// ---------------------------------------------------------------------------
__global__ __launch_bounds__(32) void conv_wmma_kernel(
    const float* __restrict__ x, const float* __restrict__ w,
    const float* __restrict__ bias, const float* __restrict__ residual,
    float* __restrict__ y,
    int Bn, int Cin, int Hin, int Win, int Cout, int Hout, int Wout,
    int KH, int KW, int stride, int pad, int transposed, int act, float scale)
{
    const int lane = threadIdx.x;
    const int hi   = lane >> 4;          // 0: lanes 0-15, 1: lanes 16-31
    const int l16  = lane & 15;
    const int KHW  = KH * KW;
    const int Ktot = Cin * KHW;
    const int HWin = Hin * Win;
    const int Npix = Bn * Hout * Wout;

    // Output pixel (B/C column) owned by this lane.
    const int p = blockIdx.x * 16 + l16;
    const bool pvalid = (p < Npix);
    int pb = 0, poh = 0, pow_ = 0;
    if (pvalid) {
        pb = p / (Hout * Wout);
        int r = p - pb * (Hout * Wout);
        poh  = r / Wout;
        pow_ = r - poh * Wout;
    }

    const int m0 = blockIdx.y * 64 + l16;                 // A row, mb=0
    const int wstride = transposed ? Cout * KHW : KHW;    // weight step per ci

    CFrag acc[4];
    #pragma unroll
    for (int mb = 0; mb < 4; ++mb)
        #pragma unroll
        for (int j = 0; j < 8; ++j) acc[mb].f[j] = 0.f;

    // ---- K loop: kernel tap outer, input channel inner (add-only inner) ----
    for (int khw = 0; khw < KHW; ++khw) {
        const int kh = khw / KW;                 // tiny outer-loop divides only
        const int kw = khw - kh * KW;

        int ih, iw; bool ok = pvalid;
        if (!transposed) {
            ih = poh  * stride - pad + kh;
            iw = pow_ * stride - pad + kw;
        } else {
            // ConvTranspose2d: oh = ih*s - p + kh  =>  ih = (oh+p-kh)/s
            int th = poh + pad - kh, tw = pow_ + pad - kw;
            if (th < 0 || tw < 0 || (th % stride) || (tw % stride)) ok = false;
            ih = th / stride; iw = tw / stride;
        }
        if (ih < 0 || ih >= Hin || iw < 0 || iw >= Win) ok = false;
        const int baseB = ((pb * Cin) * Hin + ih) * Win + iw;   // ci = 0

        long wb[4];
        #pragma unroll
        for (int mb = 0; mb < 4; ++mb) {
            int am = m0 + mb * 16;
            wb[mb] = (transposed ? (long)am * KHW : (long)am * Ktot) + khw;
        }

        for (int cc = 0; cc < Cin; cc += 32) {
            const bool full = (cc + 32 <= Cin);
            // ---- B fragment (im2col): lane-coalesced, stride HWin over K ----
            AFrag b;
            #pragma unroll
            for (int i = 0; i < 16; ++i) {
                int ci = cc + i + hi * 16;
                float bv = 0.f;
                if (ok && (full || ci < Cin)) bv = x[baseB + ci * HWin];
                b.h[i] = (_Float16)bv;
            }
            if (ok && cc + 32 < Cin)
                __builtin_prefetch(&x[baseB + (cc + 32) * HWin], 0, 1);
            // ---- 4 A fragments reuse this B fragment ----
            #pragma unroll
            for (int mb = 0; mb < 4; ++mb) {
                const int am = m0 + mb * 16;
                AFrag a;
                #pragma unroll
                for (int i = 0; i < 16; ++i) {
                    int ci = cc + i + hi * 16;
                    float av = 0.f;
                    if (am < Cout && (full || ci < Cin))
                        av = w[wb[mb] + (long)ci * wstride];
                    a.h[i] = (_Float16)av;
                }
                acc[mb].v = __builtin_amdgcn_wmma_f32_16x16x32_f16(
                    false, a.v, false, b.v, (short)0, acc[mb].v, false, false);
            }
        }
    }

    if (!pvalid) return;
    #pragma unroll
    for (int mb = 0; mb < 4; ++mb) {
        #pragma unroll
        for (int j = 0; j < 8; ++j) {
            int m = blockIdx.y * 64 + mb * 16 + j + hi * 8;
            if (m >= Cout) continue;
            float v = (acc[mb].f[j] + bias[m]) * scale;
            long oidx = (((long)pb * Cout + m) * Hout + poh) * Wout + pow_;
            if (residual) v += residual[oidx];
            if (act == 1)      v = v > 0.f ? v : 0.f;
            else if (act == 2) v = 1.f / (1.f + __expf(-v));
            y[oidx] = v;
        }
    }
}

// ---------------------------------------------------------------------------
// ||c_j||^2 for each of the 1024 codebook rows.
// ---------------------------------------------------------------------------
__global__ void cnorm_kernel(const float* __restrict__ cb, float* __restrict__ cn)
{
    int j = blockIdx.x * blockDim.x + threadIdx.x;
    if (j >= 1024) return;
    float s = 0.f;
    #pragma unroll 8
    for (int t = 0; t < 256; ++t) { float v = cb[j * 256 + t]; s += v * v; }
    cn[j] = s;
}

// ---------------------------------------------------------------------------
// Vector quantize: 25600 rows (256 contiguous floats: faithful NCHW flatten),
// 1024 codes. argmin_j(||c_j||^2 - 2 z.c_j) via WMMA. Codebook tiles staged
// into LDS by the Tensor Data Mover with padded rows (stride 260 dwords) for
// conflict-free column reads.
// ---------------------------------------------------------------------------
#define VQ_ROWPAD 260  // 256 + 4 dwords TDM pad per row

__global__ __launch_bounds__(32) void vq_wmma_kernel(
    const float* __restrict__ z, const float* __restrict__ cb,
    const float* __restrict__ cnorm, float* __restrict__ zq,
    int* __restrict__ indices)
{
    // Single static shared array => LDS byte offset 0 (used as D#.lds_addr).
    __shared__ float scode[16 * VQ_ROWPAD];

    const int lane = threadIdx.x;
    const int hi   = lane >> 4;
    const int l16  = lane & 15;
    const int tile = blockIdx.x;
    const long rbase = (long)tile * 16 * 256;

    // Preload all 8 A fragments (this wave's 16 z-rows, K=256); reused across
    // all 64 code tiles (stays resident in 64 VGPRs).
    AFrag afr[8];
    #pragma unroll
    for (int c8 = 0; c8 < 8; ++c8)
        #pragma unroll
        for (int i = 0; i < 16; ++i) {
            int kk = c8 * 32 + i + hi * 16;
            afr[c8].h[i] = (_Float16)z[rbase + (long)l16 * 256 + kk];
        }

    float bestv[8]; int besti[8];
    #pragma unroll
    for (int j = 0; j < 8; ++j) { bestv[j] = 3.4e38f; besti[j] = 0; }

    for (int ct = 0; ct < 64; ++ct) {
        const int code = ct * 16 + l16;         // this lane's column (code id)
#ifdef HAS_TDM
        __syncthreads();   // prior LDS reads complete before TDM overwrites
        {
            // D# per CDNA5 ISA ch.8: 2-D tensor, data_size=4B,
            // tile 256x16, tensor_dim0_stride=256, LDS pad: every 256 dwords
            // insert 4 dwords (pad_interval code 7, pad_amount code 3).
            unsigned long long ga =
                (unsigned long long)(size_t)(cb + (size_t)ct * 16 * 256);
            v4u g0;
            g0[0] = 1u;                                  // count=1, user D#
            g0[1] = 0u;                                  // lds_addr = 0
            g0[2] = (unsigned int)ga;                    // global_addr[31:0]
            g0[3] = (unsigned int)((ga >> 32) & 0x1FFFFFFu) | 0x80000000u; // type=2
            v8i g1;
            g1[0] = 0x07D20000;      // data_size=4B | pad_en | intv=7 | amt=3
            g1[1] = (256 & 0xFFFF) << 16;                // tensor_dim0 = 256
            g1[2] = (1024 & 0xFFFF) << 16;               // tensor_dim1 = 1024
            g1[3] = 256 << 16;                           // tile_dim0 = 256
            g1[4] = 16;                                  // tile_dim1=16, dim2=0
            g1[5] = 256;                                 // tensor_dim0_stride
            g1[6] = 0; g1[7] = 0;                        // dim1_stride unused
            v4i gz = (v4i)0;
#  if __has_include(<hip/amd_detail/amd_gfx1250_TDM.h>)
            v8i gz8 = (v8i)0;
            __builtin_amdgcn_tensor_load_to_lds(g0, g1, gz, gz, gz8, 0);
#  else
            __builtin_amdgcn_tensor_load_to_lds(g0, g1, gz, gz, 0);
#  endif
            __builtin_amdgcn_s_wait_tensorcnt(0);
        }
        __syncthreads();
#endif
        CFrag acc;
        #pragma unroll
        for (int j = 0; j < 8; ++j) acc.f[j] = 0.f;
        #pragma unroll
        for (int c8 = 0; c8 < 8; ++c8) {
            AFrag b;
            #pragma unroll
            for (int i = 0; i < 16; ++i) {
                int kk = c8 * 32 + i + hi * 16;
#ifdef HAS_TDM
                b.h[i] = (_Float16)scode[l16 * VQ_ROWPAD + kk];
#else
                b.h[i] = (_Float16)cb[(long)code * 256 + kk];
#endif
            }
            acc.v = __builtin_amdgcn_wmma_f32_16x16x32_f16(
                false, afr[c8].v, false, b.v, (short)0, acc.v, false, false);
        }
        const float cn = cnorm[code];
        #pragma unroll
        for (int j = 0; j < 8; ++j) {
            float s = cn - 2.f * acc.f[j];  // ||z||^2 constant per row: omitted
            if (s < bestv[j] || (s == bestv[j] && code < besti[j])) {
                bestv[j] = s; besti[j] = code;
            }
        }
    }

    // Min-reduce (val, idx) across the 16 lanes (columns) of each half-group.
    #pragma unroll
    for (int j = 0; j < 8; ++j) {
        float v = bestv[j]; int ix = besti[j];
        #pragma unroll
        for (int off = 8; off >= 1; off >>= 1) {
            float ov = __shfl_xor(v,  off, 32);
            int   oi = __shfl_xor(ix, off, 32);
            if (ov < v || (ov == v && oi < ix)) { v = ov; ix = oi; }
        }
        bestv[j] = v; besti[j] = ix;
    }

    int* sI = (int*)scode;     // reuse LDS for the per-row winner broadcast
    __syncthreads();
    if (l16 == 0) {
        #pragma unroll
        for (int j = 0; j < 8; ++j) sI[hi * 8 + j] = besti[j];   // row j+8*hi
    }
    __syncthreads();

    if (lane < 16) indices[tile * 16 + lane] = sI[lane];

    // zq = codebook[idx] (forward value of straight-through estimator).
    for (int t = lane; t < 16 * 64; t += 32) {
        int row = t >> 6, q = t & 63;
        const float4* src = (const float4*)(cb + (long)sI[row] * 256) + q;
        float4*       dst = (float4*)(zq + rbase + (long)row * 256) + q;
        *dst = *src;
    }
}

// ---------------------------------------------------------------------------
// Host launcher
// ---------------------------------------------------------------------------
extern "C" void kernel_launch(void* const* d_in, const int* in_sizes, int n_in,
                              void* d_out, int out_size, void* d_ws, size_t ws_size,
                              hipStream_t stream)
{
    const float* x     = (const float*)d_in[0];
    const float* w0    = (const float*)d_in[1];  const float* b0    = (const float*)d_in[2];
    const float* a_c1w = (const float*)d_in[3];  const float* a_c1b = (const float*)d_in[4];
    const float* a_c2w = (const float*)d_in[5];  const float* a_c2b = (const float*)d_in[6];
    const float* a_scw = (const float*)d_in[7];  const float* a_scb = (const float*)d_in[8];
    const float* b_c1w = (const float*)d_in[9];  const float* b_c1b = (const float*)d_in[10];
    const float* b_c2w = (const float*)d_in[11]; const float* b_c2b = (const float*)d_in[12];
    const float* b_scw = (const float*)d_in[13]; const float* b_scb = (const float*)d_in[14];
    const float* ew    = (const float*)d_in[15]; const float* eb    = (const float*)d_in[16];
    const float* cbook = (const float*)d_in[17];
    const float* d0w   = (const float*)d_in[18]; const float* d0b   = (const float*)d_in[19];
    const float* dt1w  = (const float*)d_in[20]; const float* dt1b  = (const float*)d_in[21];
    const float* dt2w  = (const float*)d_in[22]; const float* dt2b  = (const float*)d_in[23];
    const float* dow   = (const float*)d_in[24]; const float* dob   = (const float*)d_in[25];

    float* out   = (float*)d_out;
    float* recon = out;                                  // 409600
    float* zq    = out + 409600;                         // 6553600
    int*   idxo  = (int*)(out + 409600 + 6553600);       // 25600

    // Three reused scratch regions (peak live set ~420 MB):
    float* ws = (float*)d_ws;
    float* R0 = ws;                         // 52,428,800 floats
    float* R1 = ws + 52428800;              // 26,214,400 floats
    float* R2 = ws + 52428800 + 26214400;   // 26,214,400 floats
    float* cn = R2;                         // cnorm (1024) lives in R2 during VQ

    auto launch_conv = [&](const float* in, const float* wt, const float* bs,
                           const float* res, float* op,
                           int B, int Ci, int Hi, int Wi, int Co, int Ho, int Wo,
                           int KH, int KW, int st, int pd, int tr, int act, float sc) {
        int np = B * Ho * Wo;
        dim3 grid((np + 15) / 16, (Co + 63) / 64);
        conv_wmma_kernel<<<grid, 32, 0, stream>>>(in, wt, bs, res, op,
            B, Ci, Hi, Wi, Co, Ho, Wo, KH, KW, st, pd, tr, act, sc);
    };

    // ---------------- encoder ----------------
    launch_conv(x,  w0,    b0,    nullptr, R0, 4, 1,   320, 320, 128, 320, 320, 3, 3, 1, 1, 0, 1, BNF);
    launch_conv(R0, a_scw, a_scb, nullptr, R1, 4, 128, 320, 320, 256, 160, 160, 1, 1, 2, 0, 0, 0, BNF);
    launch_conv(R0, a_c1w, a_c1b, nullptr, R2, 4, 128, 320, 320, 256, 160, 160, 3, 3, 2, 1, 0, 1, BNF);
    launch_conv(R2, a_c2w, a_c2b, R1,      R0, 4, 256, 160, 160, 256, 160, 160, 3, 3, 1, 1, 0, 1, BNF);
    launch_conv(R0, b_scw, b_scb, nullptr, R1, 4, 256, 160, 160, 512, 80,  80,  1, 1, 2, 0, 0, 0, BNF);
    launch_conv(R0, b_c1w, b_c1b, nullptr, R2, 4, 256, 160, 160, 512, 80,  80,  3, 3, 2, 1, 0, 1, BNF);
    launch_conv(R2, b_c2w, b_c2b, R1,      R0, 4, 512, 80,  80,  512, 80,  80,  3, 3, 1, 1, 0, 1, BNF);
    launch_conv(R0, ew,    eb,    nullptr, R1, 4, 512, 80,  80,  256, 80,  80,  1, 1, 1, 0, 0, 0, BNF);

    // ---------------- vector quantize ----------------
    cnorm_kernel<<<4, 256, 0, stream>>>(cbook, cn);
    vq_wmma_kernel<<<1600, 32, 0, stream>>>(R1, cbook, cn, zq, idxo);

    // ---------------- decoder ----------------
    launch_conv(zq, d0w,  d0b,  nullptr, R1, 4, 256, 80,  80,  512, 80,  80,  1, 1, 1, 0, 0, 1, BNF);
    launch_conv(R1, dt1w, dt1b, nullptr, R2, 4, 512, 80,  80,  256, 160, 160, 4, 4, 2, 1, 1, 1, BNF);
    launch_conv(R2, dt2w, dt2b, nullptr, R0, 4, 256, 160, 160, 128, 320, 320, 4, 4, 2, 1, 1, 1, BNF);
    launch_conv(R0, dow,  dob,  nullptr, recon, 4, 128, 320, 320, 1, 320, 320, 3, 3, 1, 1, 0, 2, 1.0f);

    (void)in_sizes; (void)n_in; (void)out_size; (void)ws_size;
}
